// KernelizedAttention_53987738911199
// MI455X (gfx1250) — matrix operand order repaired and checked by reading, not compiled, not run on hardware
//
#include <hip/hip_runtime.h>
#include <stdint.h>
#include <stddef.h>

// ---------------------------------------------------------------------------
// KernelizedAttention for MI455X (gfx1250): fused f16-WMMA K/V projection with
// in-epilogue kv/k reductions; async LDS staging when available.
// ---------------------------------------------------------------------------

typedef __attribute__((ext_vector_type(16))) _Float16 v16h;
typedef __attribute__((ext_vector_type(8)))  _Float16 v8h;
typedef __attribute__((ext_vector_type(4)))  _Float16 v4h;
typedef __attribute__((ext_vector_type(8)))  float    v8f;

#define D_DIM   2048
#define H_DIM   16
#define B_DIM   4
#define T_DIM   4096
#define M_TOTAL (T_DIM * B_DIM)   // 16384 rows (t*B + b)
#define EPSF    1e-6f

// LDS tile geometry: 128 rows x 32 K-halves per tile, row padded to 40 halves
#define KSTEP   32
#define LROW    40
#define TILE_H  (128 * LROW)       // 5120 halves per tile
#define BUF_H   (4 * TILE_H)       // 4 tiles (Ak, Av, Bk, Bv) per buffer
#define SMEM_BYTES (2 * BUF_H * 2) // double buffered: 81920 bytes

// -------------------- async-LDS path (guarded; sync fallback) ---------------
#if defined(__HIP_DEVICE_COMPILE__) && \
    __has_builtin(__builtin_amdgcn_global_load_async_to_lds_b128) && \
    __has_builtin(__builtin_amdgcn_s_wait_asynccnt)
#define USE_ASYNC 1
#else
#define USE_ASYNC 0
#endif

#if USE_ASYNC
// Builtin prototype (from hipcc diagnostic): param0 = int __vector(4) __device__*,
// i.e. non-const 16-byte int vector pointer in global AS; LDS side in AS(3).
typedef int ka_v4i __attribute__((vector_size(16)));
typedef __attribute__((address_space(1))) ka_v4i ka_as1_v4i;
typedef __attribute__((address_space(3))) ka_v4i ka_as3_v4i;
#endif

__device__ __forceinline__ void copy16(_Float16* dst_lds, const _Float16* src_g) {
#if USE_ASYNC
  __builtin_amdgcn_global_load_async_to_lds_b128(
      (ka_as1_v4i*)src_g, (ka_as3_v4i*)dst_lds, 0, 0);
#else
  *(uint4*)dst_lds = *(const uint4*)src_g;
#endif
}

#if USE_ASYNC
#define WAIT_ASYNC(n) __builtin_amdgcn_s_wait_asynccnt(n)
#else
#define WAIT_ASYNC(n) ((void)0)
#endif

__device__ __forceinline__ void cvt8(_Float16* dst_lds, const float* src_g) {
  float4 a = *(const float4*)src_g;
  float4 b = *(const float4*)(src_g + 4);
  v8h h;
  h[0] = (_Float16)a.x; h[1] = (_Float16)a.y; h[2] = (_Float16)a.z; h[3] = (_Float16)a.w;
  h[4] = (_Float16)b.x; h[5] = (_Float16)b.y; h[6] = (_Float16)b.z; h[7] = (_Float16)b.w;
  *(v8h*)dst_lds = h;
}

// -------------------- f32 -> f16 conversion pass ----------------------------
__global__ void ka_cvt_f16(const float4* __restrict__ src, v4h* __restrict__ dst, int n4) {
  int stride = gridDim.x * blockDim.x;
  for (int i = blockIdx.x * blockDim.x + threadIdx.x; i < n4; i += stride) {
    float4 a = src[i];
    v4h h;
    h[0] = (_Float16)a.x; h[1] = (_Float16)a.y;
    h[2] = (_Float16)a.z; h[3] = (_Float16)a.w;
    dst[i] = h;
  }
}

// -------------------- zero accumulators -------------------------------------
__global__ void ka_zero(float* __restrict__ kv, float* __restrict__ ks) {
  int i = blockIdx.x * blockDim.x + threadIdx.x;
  if (i < B_DIM * D_DIM) kv[i] = 0.f;
  if (i < B_DIM * H_DIM) ks[i] = 0.f;
}

// -------------------- fused K/V projection + reduction ----------------------
// Block: 256 threads (8 waves). Tile: M 128 x N 128 (N-block == head).
// Wave w owns rows [16w,16w+16) x all 8 N-subtiles, for K and V GEMMs.
template <bool CVT>
__global__ __launch_bounds__(256)
void ka_fused_kv(const void* __restrict__ khist, const void* __restrict__ vhist,
                 const void* __restrict__ wk,    const void* __restrict__ wv,
                 const float* __restrict__ bk,   const float* __restrict__ bv,
                 float* __restrict__ kv_sum,     float* __restrict__ k_sum) {
  extern __shared__ __align__(16) unsigned char smem[];
  _Float16* sh = (_Float16*)smem;

  const int t    = threadIdx.x;
  const int wave = t >> 5;
  const int lane = t & 31;
  const int m0   = blockIdx.x * 128;
  const int n0   = blockIdx.y * 128;  // head == blockIdx.y (hd == 128)

  v8f accK[8], accV[8];
#pragma unroll
  for (int i = 0; i < 8; ++i) {
    accK[i] = (v8f){0.f, 0.f, 0.f, 0.f, 0.f, 0.f, 0.f, 0.f};
    accV[i] = (v8f){0.f, 0.f, 0.f, 0.f, 0.f, 0.f, 0.f, 0.f};
  }

  auto stage = [&](int kk, int buf) {
    _Float16* base = sh + buf * BUF_H;
#pragma unroll
    for (int i = 0; i < 2; ++i) {
      int c   = t + 256 * i;        // 512 chunks of 8 halves per tile
      int row = c >> 2;
      int seg = c & 3;
      int lo  = row * LROW + seg * 8;
      size_t ga = (size_t)(m0 + row) * D_DIM + (size_t)(kk + seg * 8);
      size_t gw = (size_t)(n0 + row) * D_DIM + (size_t)(kk + seg * 8);
      if constexpr (!CVT) {
        const _Float16* kh16 = (const _Float16*)khist;
        const _Float16* vh16 = (const _Float16*)vhist;
        const _Float16* wk16 = (const _Float16*)wk;
        const _Float16* wv16 = (const _Float16*)wv;
        copy16(base + 0 * TILE_H + lo, kh16 + ga);
        copy16(base + 1 * TILE_H + lo, vh16 + ga);
        copy16(base + 2 * TILE_H + lo, wk16 + gw);
        copy16(base + 3 * TILE_H + lo, wv16 + gw);
      } else {
        const float* khf = (const float*)khist;
        const float* vhf = (const float*)vhist;
        const float* wkf = (const float*)wk;
        const float* wvf = (const float*)wv;
        cvt8(base + 0 * TILE_H + lo, khf + ga);
        cvt8(base + 1 * TILE_H + lo, vhf + ga);
        cvt8(base + 2 * TILE_H + lo, wkf + gw);
        cvt8(base + 3 * TILE_H + lo, wvf + gw);
      }
    }
  };

  const int NSTEP = D_DIM / KSTEP;  // 64
  const int ml = lane & 15;
  const int hf = lane >> 4;

  stage(0, 0);
  for (int s = 0; s < NSTEP; ++s) {
    const int cur = s & 1;
    if (s + 1 < NSTEP) {
      stage((s + 1) * KSTEP, cur ^ 1);
      WAIT_ASYNC(8);    // in-order: all but the 8 just-issued copies are done
    } else {
      WAIT_ASYNC(0);
    }
    __syncthreads();    // buffer `cur` visible to all waves

    const _Float16* base = sh + cur * BUF_H;
    const _Float16* Ak = base;
    const _Float16* Av = base + 1 * TILE_H;
    const _Float16* Bk = base + 2 * TILE_H;
    const _Float16* Bv = base + 3 * TILE_H;

    // A fragment (ISA 16-bit 16x32 layout): lane m holds K {hf*8..+7, 16+hf*8..+7}
    const _Float16* aK = Ak + (wave * 16 + ml) * LROW;
    const _Float16* aV = Av + (wave * 16 + ml) * LROW;
    v16h fak, fav;
    ((uint4*)&fak)[0] = *(const uint4*)(aK + hf * 8);
    ((uint4*)&fak)[1] = *(const uint4*)(aK + 16 + hf * 8);
    ((uint4*)&fav)[0] = *(const uint4*)(aV + hf * 8);
    ((uint4*)&fav)[1] = *(const uint4*)(aV + 16 + hf * 8);

#pragma unroll
    for (int ns = 0; ns < 8; ++ns) {
      // B fragment: lane n = ml holds K {hf*8..+7, 16+hf*8..+7} (matches A)
      const _Float16* pBk = Bk + (ns * 16 + ml) * LROW + hf * 8;
      const _Float16* pBv = Bv + (ns * 16 + ml) * LROW + hf * 8;
      v16h fbk, fbv;
      ((uint4*)&fbk)[0] = *(const uint4*)(pBk);
      ((uint4*)&fbk)[1] = *(const uint4*)(pBk + 16);
      ((uint4*)&fbv)[0] = *(const uint4*)(pBv);
      ((uint4*)&fbv)[1] = *(const uint4*)(pBv + 16);
      accK[ns] = __builtin_amdgcn_wmma_f32_16x16x32_f16(
          false, fak, false, fbk, (short)0, accK[ns], false, false);
      accV[ns] = __builtin_amdgcn_wmma_f32_16x16x32_f16(
          false, fav, false, fbv, (short)0, accV[ns], false, false);
    }
    __syncthreads();    // done reading `cur` before it is overwritten
  }

  // ----- epilogue: bias, relu+eps, k*v, reduce over rows, atomics -----------
  float* lds_kv  = (float*)smem;                 // [4][128]
  float* lds_ks  = (float*)(smem + 4096);        // [4][128]
  float* lds_ks4 = (float*)(smem + 8192);        // [4]
  for (int i = t; i < 512; i += 256) { lds_kv[i] = 0.f; lds_ks[i] = 0.f; }
  if (t < 4) lds_ks4[t] = 0.f;
  __syncthreads();

#pragma unroll
  for (int ns = 0; ns < 8; ++ns) {
    int nl = ns * 16 + ml;
    int ng = n0 + nl;
    float bkv = bk[ng];
    float bvv = bv[ng];
    float pkv[4] = {0.f, 0.f, 0.f, 0.f};
    float pks[4] = {0.f, 0.f, 0.f, 0.f};
#pragma unroll
    for (int r = 0; r < 8; ++r) {
      // D-matrix: VGPR r holds M = r + hf*8 ; global row % 4 == r % 4
      float kk = accK[ns][r] + bkv;
      kk = fmaxf(kk, 0.f) + EPSF;
      float vv = accV[ns][r] + bvv;
      int b = r & 3;
      pkv[b] += kk * vv;
      pks[b] += kk;
    }
#pragma unroll
    for (int b = 0; b < 4; ++b) {
      atomicAdd(&lds_kv[b * 128 + nl], pkv[b]);
      atomicAdd(&lds_ks[b * 128 + nl], pks[b]);
    }
  }
  __syncthreads();

  for (int i = t; i < 512; i += 256) {
    int b  = i >> 7;
    int nl = i & 127;
    atomicAdd(&kv_sum[b * D_DIM + n0 + nl], lds_kv[i]);
    atomicAdd(&lds_ks4[b], lds_ks[i]);
  }
  __syncthreads();
  if (t < 4) atomicAdd(&k_sum[t * H_DIM + blockIdx.y], lds_ks4[t]);
}

// -------------------- wave-per-output dot kernels ---------------------------
__device__ __forceinline__ float ka_wave_reduce(float s) {
#pragma unroll
  for (int off = 16; off > 0; off >>= 1) s += __shfl_down(s, off, 32);
  return s;
}

__global__ __launch_bounds__(256)
void ka_qk_combine(const float* __restrict__ q,  const float* __restrict__ Wq,
                   const float* __restrict__ bq, const float* __restrict__ kv,
                   const float* __restrict__ ks, const float* __restrict__ alpha,
                   float* __restrict__ outb) {
  const int lane = threadIdx.x & 31;
  const int idx  = blockIdx.x * 8 + (threadIdx.x >> 5);   // (b, n) output
  const int b = idx >> 11;
  const int n = idx & (D_DIM - 1);
  const float4* qr = (const float4*)(q  + (size_t)b * D_DIM);
  const float4* wr = (const float4*)(Wq + (size_t)n * D_DIM);
  float s = 0.f;
  for (int i = lane; i < D_DIM / 4; i += 32) {
    float4 a = qr[i], w = wr[i];
    s += a.x * w.x + a.y * w.y + a.z * w.z + a.w * w.w;
  }
  s = ka_wave_reduce(s);
  if (lane == 0) {
    s += bq[n];
    s = fmaxf(s, 0.f) + EPSF;                 // kernel_fn
    int h = n >> 7;
    float denom = ks[b * H_DIM + h] + EPSF;
    outb[idx] = s * (kv[b * D_DIM + n] / denom) * alpha[h];
  }
}

__global__ __launch_bounds__(256)
void ka_out_proj(const float* __restrict__ outb, const float* __restrict__ Wo,
                 const float* __restrict__ bo,   float* __restrict__ y) {
  const int lane = threadIdx.x & 31;
  const int idx  = blockIdx.x * 8 + (threadIdx.x >> 5);
  const int b = idx >> 11;
  const int n = idx & (D_DIM - 1);
  const float4* xr = (const float4*)(outb + (size_t)b * D_DIM);
  const float4* wr = (const float4*)(Wo   + (size_t)n * D_DIM);
  float s = 0.f;
  for (int i = lane; i < D_DIM / 4; i += 32) {
    float4 a = xr[i], w = wr[i];
    s += a.x * w.x + a.y * w.y + a.z * w.z + a.w * w.w;
  }
  s = ka_wave_reduce(s);
  if (lane == 0) y[idx] = s + bo[n];
}

// -------------------- host launcher -----------------------------------------
extern "C" void kernel_launch(void* const* d_in, const int* in_sizes, int n_in,
                              void* d_out, int out_size, void* d_ws, size_t ws_size,
                              hipStream_t stream) {
  (void)in_sizes; (void)n_in; (void)out_size;
  const float* q     = (const float*)d_in[0];
  const float* kh    = (const float*)d_in[1];
  const float* vh    = (const float*)d_in[2];
  const float* Wq    = (const float*)d_in[3];
  const float* bq    = (const float*)d_in[4];
  const float* Wk    = (const float*)d_in[5];
  const float* bk    = (const float*)d_in[6];
  const float* Wv    = (const float*)d_in[7];
  const float* bv    = (const float*)d_in[8];
  const float* Wo    = (const float*)d_in[9];
  const float* bo    = (const float*)d_in[10];
  const float* alpha = (const float*)d_in[11];
  float* y = (float*)d_out;

  // workspace layout
  const size_t WS_KV   = 0;                                   // 4*2048 f32
  const size_t WS_KS   = 32768;                               // 4*16 f32
  const size_t WS_OUTB = 33024;                               // 4*2048 f32
  const size_t WS_F16  = 65536;
  const size_t KH16_SZ = (size_t)M_TOTAL * D_DIM * 2;
  const size_t W16_SZ  = (size_t)D_DIM * D_DIM * 2;
  const size_t KH16_O  = WS_F16;
  const size_t VH16_O  = KH16_O + KH16_SZ;
  const size_t WK16_O  = VH16_O + KH16_SZ;
  const size_t WV16_O  = WK16_O + W16_SZ;
  const size_t WS_END  = WV16_O + W16_SZ;

  char* ws = (char*)d_ws;
  float* kv   = (float*)(ws + WS_KV);
  float* ks   = (float*)(ws + WS_KS);
  float* outb = (float*)(ws + WS_OUTB);

  ka_zero<<<(B_DIM * D_DIM + 255) / 256, 256, 0, stream>>>(kv, ks);

  dim3 grid(M_TOTAL / 128, D_DIM / 128);
  if (ws_size >= WS_END) {
    _Float16* kh16 = (_Float16*)(ws + KH16_O);
    _Float16* vh16 = (_Float16*)(ws + VH16_O);
    _Float16* wk16 = (_Float16*)(ws + WK16_O);
    _Float16* wv16 = (_Float16*)(ws + WV16_O);
    ka_cvt_f16<<<4096, 256, 0, stream>>>((const float4*)kh, (v4h*)kh16, M_TOTAL * D_DIM / 4);
    ka_cvt_f16<<<4096, 256, 0, stream>>>((const float4*)vh, (v4h*)vh16, M_TOTAL * D_DIM / 4);
    ka_cvt_f16<<<1024, 256, 0, stream>>>((const float4*)Wk, (v4h*)wk16, D_DIM * D_DIM / 4);
    ka_cvt_f16<<<1024, 256, 0, stream>>>((const float4*)Wv, (v4h*)wv16, D_DIM * D_DIM / 4);
    ka_fused_kv<false><<<grid, 256, SMEM_BYTES, stream>>>(kh16, vh16, wk16, wv16,
                                                          bk, bv, kv, ks);
  } else {
    // workspace too small for f16 mirror: convert on the fly inside the kernel
    ka_fused_kv<true><<<grid, 256, SMEM_BYTES, stream>>>(kh, vh, Wk, Wv,
                                                         bk, bv, kv, ks);
  }

  ka_qk_combine<<<(B_DIM * D_DIM) / 8, 256, 0, stream>>>(q, Wq, bq, kv, ks, alpha, outb);
  ka_out_proj<<<(B_DIM * D_DIM) / 8, 256, 0, stream>>>(outb, Wo, bo, y);
}
